// FasterRCNNMobileNet_67877663146183
// MI455X (gfx1250) — matrix operand order repaired and checked by reading, not compiled, not run on hardware
//
#include <hip/hip_runtime.h>
#include <math.h>

typedef __attribute__((ext_vector_type(16))) _Float16 v16h;
typedef __attribute__((ext_vector_type(8)))  _Float16 v8h;
typedef __attribute__((ext_vector_type(8)))  float    v8f;

#define IMGF   800.0f
#define CLIPF  4.135166556742356f   // log(1000/16)
#define NPOS   13125                 // 100*100 + 50*50 + 25*25
#define NANCH  196875                // NPOS * 15
#define PRE    500
#define CAND   1500
#define POST   512
#define DETS   100
#define FCK    12544                 // 256*7*7
#define HEADN  96                    // 15 cls + 60 box, padded to 96
#define MROWS  26250                 // 2*NPOS
#define MPAD32 26272                 // ceil(MROWS/32)*32

// ---------------------------------------------------------------- helpers
__device__ __forceinline__ float iou_f(float ax1,float ay1,float ax2,float ay2,
                                       float bx1,float by1,float bx2,float by2){
  float aa=(ax2-ax1)*(ay2-ay1);
  float ab=(bx2-bx1)*(by2-by1);
  float lx=fmaxf(ax1,bx1), ly=fmaxf(ay1,by1);
  float rx=fminf(ax2,bx2), ry=fminf(ay2,by2);
  float w=fmaxf(rx-lx,0.f), h=fmaxf(ry-ly,0.f);
  float inter=w*h;
  return inter/(aa+ab-inter+1e-9f);
}

__device__ __forceinline__ v16h cat8(v8h lo, v8h hi){
  return __builtin_shufflevector(lo, hi, 0,1,2,3,4,5,6,7,8,9,10,11,12,13,14,15);
}

// ------------------------------------------------- weight conversion kernels
__global__ void k_cvt_f16(const float* __restrict__ in, _Float16* __restrict__ out, int n){
  int i = blockIdx.x*blockDim.x + threadIdx.x;
  if (i < n) out[i] = (_Float16)in[i];
}

// transpose-convert: out[n*K + k] = in[k*N + n]   (build B^T, K-contiguous rows)
__global__ void k_cvt_t(const float* __restrict__ in, _Float16* __restrict__ out, int K, int N){
  int i = blockIdx.x*blockDim.x + threadIdx.x;
  if (i < K*N){
    int n = i / K, k = i - n*K;
    out[i] = (_Float16)in[(size_t)k*N + n];
  }
}

// build heads B^T [96][256]: rows 0..14 cls_w, 15..74 box_w, 75..95 zero
__global__ void k_build_headw(const float* __restrict__ cw, const float* __restrict__ bw,
                              _Float16* __restrict__ out){
  int i = blockIdx.x*blockDim.x + threadIdx.x;
  if (i < HEADN*256){
    int n = i >> 8, k = i & 255;
    float v = 0.f;
    if (n < 15) v = cw[n*256 + k];
    else if (n < 75) v = bw[(n-15)*256 + k];
    out[i] = (_Float16)v;
  }
}

// ------------------------------------------------- im2col (K-contiguous rows)
// Bcol[(b*HWpad + p)][cin*9 + (dy+1)*3 + (dx+1)] = feat[b][cin][y+dy][x+dx] (0 pad)
__global__ void k_im2col(const float* __restrict__ feat, _Float16* __restrict__ Bcol,
                         int S, int HWpad){
  const int HW = S*S;
  const int total = 2*HWpad*256;
  int i = blockIdx.x*blockDim.x + threadIdx.x;
  if (i >= total) return;
  const int cin  = i & 255;
  const int ppad = (i >> 8) % HWpad;
  const int b    = (i >> 8) / HWpad;
  int p = ppad; if (p >= HW) p = HW - 1;
  const int y = p / S, x = p - y*S;
  const float* fr = feat + ((size_t)b*256 + cin) * (size_t)HW;
  _Float16* dst = Bcol + ((size_t)b*HWpad + ppad)*2304 + cin*9;
#pragma unroll
  for (int dy = -1; dy <= 1; ++dy){
#pragma unroll
    for (int dx = -1; dx <= 1; ++dx){
      const int sy = y + dy, sx = x + dx;
      float v = (sy >= 0 && sy < S && sx >= 0 && sx < S) ? fr[sy*S + sx] : 0.f;
      dst[(dy+1)*3 + (dx+1)] = (_Float16)v;
    }
  }
}

// ------------------------------------------------- unified NT WMMA GEMM, 32x32/wave
// A [M,K] row-major f16, Bt [N,K] row-major f16.  Wave computes 32(m) x 32(n):
// 4 accumulators, 8x b128 loads per 4 WMMAs.  grid(N/32, ceil(M/32), batches), block(32).
// mode 0: C16[m*N+n] = relu(acc + biasN[n])                       (FC)
// mode 1: t16[(b*NPOS+posOff+p)*256+m] = relu(acc + biasM[m]), p<S_HW (conv, transposed store)
// mode 2: C32[m*N+n] = acc, m < Mreal                             (RPN heads)
__global__ void k_gemm_nt(const _Float16* __restrict__ A, const _Float16* __restrict__ Bt,
                          const float* __restrict__ bias,
                          _Float16* __restrict__ C16, float* __restrict__ C32,
                          int Mreal, int N, int K, int mode, int S_HW, int posOff){
  const int lane  = threadIdx.x;
  const int laneM = lane & 15;
  const int hi    = lane >> 4;
  const int m0 = blockIdx.y * 32;
  const int n0 = blockIdx.x * 32;

  const _Float16* Btb = Bt + (mode == 1 ? (size_t)blockIdx.z * (size_t)N * K : 0);
  int ar0 = m0 + laneM;      if (ar0 >= Mreal) ar0 = Mreal - 1;
  int ar1 = m0 + 16 + laneM; if (ar1 >= Mreal) ar1 = Mreal - 1;
  const _Float16* A0 = A   + (size_t)ar0 * K;
  const _Float16* A1 = A   + (size_t)ar1 * K;
  const _Float16* B0 = Btb + (size_t)(n0 + laneM) * K;
  const _Float16* B1 = Btb + (size_t)(n0 + 16 + laneM) * K;

  v8f acc00 = {}, acc01 = {}, acc10 = {}, acc11 = {};
#pragma unroll 2
  for (int k0 = 0; k0 < K; k0 += 32){
    v16h a0 = cat8(*(const v8h*)(A0 + k0 + (hi << 3)),
                   *(const v8h*)(A0 + k0 + 16 + (hi << 3)));
    v16h a1 = cat8(*(const v8h*)(A1 + k0 + (hi << 3)),
                   *(const v8h*)(A1 + k0 + 16 + (hi << 3)));
    v16h b0 = cat8(*(const v8h*)(B0 + k0 + (hi << 3)),
                   *(const v8h*)(B0 + k0 + 16 + (hi << 3)));
    v16h b1 = cat8(*(const v8h*)(B1 + k0 + (hi << 3)),
                   *(const v8h*)(B1 + k0 + 16 + (hi << 3)));
    acc00 = __builtin_amdgcn_wmma_f32_16x16x32_f16(false, a0, false, b0, (short)0, acc00, false, false);
    acc01 = __builtin_amdgcn_wmma_f32_16x16x32_f16(false, a0, false, b1, (short)0, acc01, false, false);
    acc10 = __builtin_amdgcn_wmma_f32_16x16x32_f16(false, a1, false, b0, (short)0, acc10, false, false);
    acc11 = __builtin_amdgcn_wmma_f32_16x16x32_f16(false, a1, false, b1, (short)0, acc11, false, false);
  }

  if (mode == 0){
#pragma unroll
    for (int r = 0; r < 8; ++r){
      const int mA = m0 + (hi << 3) + r;
      const int mB = mA + 16;
      const int nA = n0 + laneM, nB = nA + 16;
      float v00 = acc00[r] + bias[nA];
      float v01 = acc01[r] + bias[nB];
      float v10 = acc10[r] + bias[nA];
      float v11 = acc11[r] + bias[nB];
      if (v00 < 0.f) v00 = 0.f;
      if (v01 < 0.f) v01 = 0.f;
      if (v10 < 0.f) v10 = 0.f;
      if (v11 < 0.f) v11 = 0.f;
      C16[(size_t)mA * N + nA] = (_Float16)v00;
      C16[(size_t)mA * N + nB] = (_Float16)v01;
      C16[(size_t)mB * N + nA] = (_Float16)v10;
      C16[(size_t)mB * N + nB] = (_Float16)v11;
    }
  } else if (mode == 1){
    v8h h00, h01, h10, h11;
#pragma unroll
    for (int r = 0; r < 8; ++r){
      const int mA = m0 + (hi << 3) + r;
      const int mB = mA + 16;
      float v00 = acc00[r] + bias[mA];
      float v01 = acc01[r] + bias[mA];
      float v10 = acc10[r] + bias[mB];
      float v11 = acc11[r] + bias[mB];
      h00[r] = (_Float16)(v00 > 0.f ? v00 : 0.f);
      h01[r] = (_Float16)(v01 > 0.f ? v01 : 0.f);
      h10[r] = (_Float16)(v10 > 0.f ? v10 : 0.f);
      h11[r] = (_Float16)(v11 > 0.f ? v11 : 0.f);
    }
    const int p0 = n0 + laneM, p1 = p0 + 16;
    const size_t rowb = (size_t)blockIdx.z * NPOS + posOff;
    if (p0 < S_HW){
      *(v8h*)(C16 + (rowb + p0)*256 + m0 + (hi << 3))      = h00;
      *(v8h*)(C16 + (rowb + p0)*256 + m0 + 16 + (hi << 3)) = h10;
    }
    if (p1 < S_HW){
      *(v8h*)(C16 + (rowb + p1)*256 + m0 + (hi << 3))      = h01;
      *(v8h*)(C16 + (rowb + p1)*256 + m0 + 16 + (hi << 3)) = h11;
    }
  } else {
#pragma unroll
    for (int r = 0; r < 8; ++r){
      const int mA = m0 + (hi << 3) + r;
      const int mB = mA + 16;
      const int nA = n0 + laneM, nB = nA + 16;
      if (mA < Mreal){
        C32[(size_t)mA * N + nA] = acc00[r];
        C32[(size_t)mA * N + nB] = acc01[r];
      }
      if (mB < Mreal){
        C32[(size_t)mB * N + nA] = acc10[r];
        C32[(size_t)mB * N + nB] = acc11[r];
      }
    }
  }
}

// ------------------------------------------------- scatter heads output + bias
__global__ void k_split_heads(const float* __restrict__ C, const float* __restrict__ cb,
                              const float* __restrict__ bb,
                              float* __restrict__ objs, float* __restrict__ dels){
  const int total = MROWS * 75;
  int i = blockIdx.x*blockDim.x + threadIdx.x;
  if (i >= total) return;
  const int row = i / 75;
  const int j   = i - row*75;
  const int b = row / NPOS;
  const int g = row - b*NPOS;
  int lvl, p;
  if (g < 10000){ lvl = 0; p = g; }
  else if (g < 12500){ lvl = 1; p = g - 10000; }
  else { lvl = 2; p = g - 12500; }
  const int anchOff[3] = {0, 150000, 187500};
  const size_t base = (size_t)b*NANCH + anchOff[lvl] + (size_t)p*15;
  const float v = C[(size_t)row*HEADN + j];
  if (j < 15) objs[base + j] = v + cb[j];
  else {
    const int j2 = j - 15;
    dels[base*4 + j2] = v + bb[j2];
  }
}

// ------------------------------------------------- per-level top-500 + decode
__global__ void k_topk_decode(const float* __restrict__ objs, const float* __restrict__ dels,
                              float* __restrict__ scratch,
                              float* __restrict__ cand_score, float* __restrict__ cand_box){
  const int lvl = blockIdx.x, b = blockIdx.y, tid = threadIdx.x;
  const int sizes[3]   = {100, 50, 25};
  const int anchOff[3] = {0, 150000, 187500};
  const int Nl[3]      = {150000, 37500, 9375};
  const float strides[3] = {8.f, 16.f, 32.f};
  const int N = Nl[lvl];
  const size_t base = (size_t)b*NANCH + anchOff[lvl];

  __shared__ float sv[1024]; __shared__ int si[1024];
  __shared__ float selS[PRE]; __shared__ int selI[PRE];

  for (int i = tid; i < N; i += 1024) scratch[base + i] = objs[base + i];
  __syncthreads();

  for (int it = 0; it < PRE; ++it){
    float bv = -INFINITY; int bi = 0x7fffffff;
    for (int i = tid; i < N; i += 1024){
      const float v = scratch[base + i];
      if (v > bv){ bv = v; bi = i; }
    }
    sv[tid] = bv; si[tid] = bi; __syncthreads();
    for (int s = 512; s > 0; s >>= 1){
      if (tid < s){
        if (sv[tid+s] > sv[tid] || (sv[tid+s] == sv[tid] && si[tid+s] < si[tid])){
          sv[tid] = sv[tid+s]; si[tid] = si[tid+s];
        }
      }
      __syncthreads();
    }
    if (tid == 0){ selS[it] = sv[0]; selI[it] = si[0]; scratch[base + si[0]] = -INFINITY; }
    __syncthreads();
  }

  if (tid < PRE){
    const int idx = selI[tid];
    const float s = selS[tid];
    const int S = sizes[lvl];
    const float stride = strides[lvl];
    const int p = idx / 15, a = idx - p*15;
    const int iy = p / S, ix = p - iy*S;
    const int ri = a / 5, sc = a - ri*5;
    const float ratios[3] = {0.5f, 1.0f, 2.0f};
    const float scales[5] = {32.f, 64.f, 128.f, 256.f, 512.f};
    const float hr = sqrtf(ratios[ri]), wr = 1.f/hr;
    const float aw = wr*scales[sc], ah = hr*scales[sc];
    const float cx = ix*stride, cy = iy*stride;
    const float* d = dels + (base + idx)*4;
    const float dx = d[0], dy = d[1];
    const float dw = fminf(d[2], CLIPF), dh = fminf(d[3], CLIPF);
    const float pcx = dx*aw + cx, pcy = dy*ah + cy;
    const float pw = expf(dw)*aw, ph = expf(dh)*ah;
    float x1 = fminf(fmaxf(pcx - 0.5f*pw, 0.f), IMGF);
    float y1 = fminf(fmaxf(pcy - 0.5f*ph, 0.f), IMGF);
    float x2 = fminf(fmaxf(pcx + 0.5f*pw, 0.f), IMGF);
    float y2 = fminf(fmaxf(pcy + 0.5f*ph, 0.f), IMGF);
    const bool valid = (x2-x1 >= 1e-3f) && (y2-y1 >= 1e-3f);
    const int ci = b*CAND + lvl*PRE + tid;
    cand_score[ci] = valid ? s : -1e9f;
    cand_box[ci*4+0] = x1; cand_box[ci*4+1] = y1;
    cand_box[ci*4+2] = x2; cand_box[ci*4+3] = y2;
  }
}

// ------------------------------------------------- RPN sort + batched NMS + compaction
__global__ void k_rpn_nms(const float* __restrict__ cand_score, const float* __restrict__ cand_box,
                          float* __restrict__ props){
  const int b = blockIdx.x, tid = threadIdx.x, bd = blockDim.x;
  __shared__ float key[CAND];
  __shared__ float bx[CAND*4];
  __shared__ int order[CAND];
  __shared__ int keep[CAND];
  __shared__ int keeporig[CAND];

  for (int i = tid; i < CAND; i += bd){
    key[i] = cand_score[b*CAND + i];
    const float off = (float)(i / PRE) * (IMGF + 16.f);
    bx[i*4+0] = cand_box[(b*CAND+i)*4+0] + off;
    bx[i*4+1] = cand_box[(b*CAND+i)*4+1] + off;
    bx[i*4+2] = cand_box[(b*CAND+i)*4+2] + off;
    bx[i*4+3] = cand_box[(b*CAND+i)*4+3] + off;
  }
  __syncthreads();

  for (int i = tid; i < CAND; i += bd){
    const float ki = key[i];
    int r = 0;
    for (int j = 0; j < CAND; ++j){
      const float kj = key[j];
      r += (kj > ki) || (kj == ki && j < i);
    }
    order[r] = i;
    keep[i] = 1;
  }
  __syncthreads();

  for (int i = 0; i < CAND; ++i){
    __syncthreads();
    if (!keep[i]) continue;
    const int oi = order[i];
    const float ax1 = bx[oi*4], ay1 = bx[oi*4+1], ax2 = bx[oi*4+2], ay2 = bx[oi*4+3];
    for (int j = i + 1 + tid; j < CAND; j += bd){
      if (!keep[j]) continue;
      const int oj = order[j];
      if (iou_f(ax1,ay1,ax2,ay2, bx[oj*4],bx[oj*4+1],bx[oj*4+2],bx[oj*4+3]) > 0.7f)
        keep[j] = 0;
    }
  }
  __syncthreads();
  for (int i = tid; i < CAND; i += bd) keeporig[order[i]] = keep[i];
  __syncthreads();

  if (tid == 0){
    const float* cb = cand_box + (size_t)b*CAND*4;
    float* pr = props + (size_t)b*POST*4;
    int cnt = 0;
    for (int i = 0; i < CAND && cnt < POST; ++i){
      if (keep[i]){
        const int oi = order[i];
        pr[cnt*4+0]=cb[oi*4+0]; pr[cnt*4+1]=cb[oi*4+1];
        pr[cnt*4+2]=cb[oi*4+2]; pr[cnt*4+3]=cb[oi*4+3];
        ++cnt;
      }
    }
    for (int oi = 0; oi < CAND && cnt < POST; ++oi){
      if (!keeporig[oi]){
        pr[cnt*4+0]=cb[oi*4+0]; pr[cnt*4+1]=cb[oi*4+1];
        pr[cnt*4+2]=cb[oi*4+2]; pr[cnt*4+3]=cb[oi*4+3];
        ++cnt;
      }
    }
  }
}

// ------------------------------------------------- ROI align -> x16 [1024][12544] f16
__global__ void k_roi(const float* __restrict__ f0, const float* __restrict__ f1,
                      const float* __restrict__ f2, const float* __restrict__ props,
                      _Float16* __restrict__ x16){
  const long long total = 2LL*POST*256*49;
  for (long long t = blockIdx.x*(long long)blockDim.x + threadIdx.x; t < total;
       t += (long long)gridDim.x*blockDim.x){
    const int cell = (int)(t % 49);
    const int c    = (int)((t / 49) % 256);
    const int n    = (int)((t / (49*256)) % POST);
    const int b    = (int)(t / (49LL*256*POST));
    const float* pr = props + ((size_t)b*POST + n)*4;
    const float x1 = pr[0], y1 = pr[1], x2 = pr[2], y2 = pr[3];
    const float area = fmaxf(x2-x1, 0.f) * fmaxf(y2-y1, 0.f);
    float kf = floorf(4.f + log2f(sqrtf(area)/224.f + 1e-6f));
    kf = fminf(fmaxf(kf, 3.f), 5.f);
    const int lvl = (int)kf - 3;
    const int   Si[3]   = {100, 50, 25};
    const float strd[3] = {8.f, 16.f, 32.f};
    const int   S  = Si[lvl];
    const float hw = (float)S;
    const float sc = 1.f / strd[lvl];
    const float rx1 = x1*sc, ry1 = y1*sc;
    const float rw = fmaxf(x2*sc - rx1, 1.f);
    const float rh = fmaxf(y2*sc - ry1, 1.f);
    const float bw = rw / 7.f, bh = rh / 7.f;
    const int py = cell / 7, px = cell - py*7;
    const float* f = (lvl == 0) ? f0 : (lvl == 1) ? f1 : f2;
    const float* fc = f + ((size_t)b*256 + c) * (size_t)S * S;
    const int hiI = S - 1;
    float acc = 0.f;
#pragma unroll
    for (int sy = 0; sy < 2; ++sy){
#pragma unroll
      for (int sx = 0; sx < 2; ++sx){
        float Y = ry1 + ((float)py + ((float)sy + 0.5f)*0.5f) * bh;
        float X = rx1 + ((float)px + ((float)sx + 0.5f)*0.5f) * bw;
        Y = fminf(fmaxf(Y, 0.f), hw - 1.f);
        X = fminf(fmaxf(X, 0.f), hw - 1.f);
        const float y0 = floorf(Y), x0 = floorf(X);
        const float ly = Y - y0, lx = X - x0;
        const int y0i = (int)y0, x0i = (int)x0;
        const int y1i = min(y0i + 1, hiI), x1i = min(x0i + 1, hiI);
        acc += (1.f-ly)*(1.f-lx) * fc[y0i*S + x0i]
             + (1.f-ly)*lx       * fc[y0i*S + x1i]
             + ly*(1.f-lx)       * fc[y1i*S + x0i]
             + ly*lx             * fc[y1i*S + x1i];
      }
    }
    x16[((size_t)b*POST + n)*FCK + c*49 + cell] = (_Float16)(acc * 0.25f);
  }
}

// ------------------------------------------------- final cls/box heads (tiny)
__global__ void k_cls_box(const _Float16* __restrict__ h2,
                          const float* __restrict__ cw, const float* __restrict__ cb,
                          const float* __restrict__ bw, const float* __restrict__ bb,
                          float* __restrict__ cls, float* __restrict__ dlt){
  const int row = blockIdx.x*blockDim.x + threadIdx.x;
  if (row >= 1024) return;
  const _Float16* hrow = h2 + (size_t)row * 1024;
  float c[2] = {0.f, 0.f};
  float bx[8];
#pragma unroll
  for (int j = 0; j < 8; ++j) bx[j] = 0.f;
  for (int k = 0; k < 1024; ++k){
    const float h = (float)hrow[k];
    c[0] += h * cw[k*2 + 0];
    c[1] += h * cw[k*2 + 1];
#pragma unroll
    for (int j = 0; j < 8; ++j) bx[j] += h * bw[k*8 + j];
  }
  cls[row*2+0] = c[0] + cb[0];
  cls[row*2+1] = c[1] + cb[1];
#pragma unroll
  for (int j = 0; j < 8; ++j) dlt[row*8 + j] = bx[j] + bb[j];
}

// ------------------------------------------------- postprocess
__global__ void k_post(const float* __restrict__ cls, const float* __restrict__ dlt,
                       const float* __restrict__ props, float* __restrict__ out){
  const int b = blockIdx.x, tid = threadIdx.x;
  __shared__ float sc[POST];
  __shared__ float key[POST];
  __shared__ float bxx[POST*4];
  __shared__ int order[POST];
  __shared__ int keep[POST];
  __shared__ int kv[POST];

  {
    const int n = tid;
    const float c0 = cls[((size_t)b*POST + n)*2 + 0];
    const float c1 = cls[((size_t)b*POST + n)*2 + 1];
    const float mx = fmaxf(c0, c1);
    const float e0 = expf(c0 - mx), e1 = expf(c1 - mx);
    const float s = e1 / (e0 + e1);
    const float* pr = props + ((size_t)b*POST + n)*4;
    const float w = pr[2]-pr[0], h = pr[3]-pr[1];
    const float cx = pr[0] + 0.5f*w, cy = pr[1] + 0.5f*h;
    const float* d = dlt + ((size_t)b*POST + n)*8 + 4;
    const float dx = d[0]/10.f, dy = d[1]/10.f;
    const float dw = fminf(d[2]/5.f, CLIPF), dh = fminf(d[3]/5.f, CLIPF);
    const float pcx = dx*w + cx, pcy = dy*h + cy;
    const float pw = expf(dw)*w, ph = expf(dh)*h;
    float x1 = fminf(fmaxf(pcx - 0.5f*pw, 0.f), IMGF);
    float y1 = fminf(fmaxf(pcy - 0.5f*ph, 0.f), IMGF);
    float x2 = fminf(fmaxf(pcx + 0.5f*pw, 0.f), IMGF);
    float y2 = fminf(fmaxf(pcy + 0.5f*ph, 0.f), IMGF);
    const bool valid = (s > 0.05f) && (x2-x1 >= 1e-2f) && (y2-y1 >= 1e-2f);
    sc[n] = s; kv[n] = valid ? 1 : 0;
    key[n] = valid ? s : -1e9f;
    bxx[n*4+0]=x1; bxx[n*4+1]=y1; bxx[n*4+2]=x2; bxx[n*4+3]=y2;
  }
  __syncthreads();

  {
    const int i = tid;
    const float ki = key[i];
    int r = 0;
    for (int j = 0; j < POST; ++j){
      const float kj = key[j];
      r += (kj > ki) || (kj == ki && j < i);
    }
    order[r] = i;
    keep[i] = 1;
  }
  __syncthreads();

  for (int i = 0; i < POST; ++i){
    __syncthreads();
    if (!keep[i]) continue;
    const int oi = order[i];
    const float ax1=bxx[oi*4], ay1=bxx[oi*4+1], ax2=bxx[oi*4+2], ay2=bxx[oi*4+3];
    for (int j = i + 1 + tid; j < POST; j += blockDim.x){
      if (!keep[j]) continue;
      const int oj = order[j];
      if (iou_f(ax1,ay1,ax2,ay2, bxx[oj*4],bxx[oj*4+1],bxx[oj*4+2],bxx[oj*4+3]) > 0.5f)
        keep[j] = 0;
    }
  }
  __syncthreads();

  if (tid == 0){
    float* o = out + (size_t)b*DETS*6;
    int cnt = 0;
    for (int i = 0; i < POST && cnt < DETS; ++i){
      const int oi = order[i];
      if (keep[i] && kv[oi]){
        o[cnt*6+0]=bxx[oi*4+0]; o[cnt*6+1]=bxx[oi*4+1];
        o[cnt*6+2]=bxx[oi*4+2]; o[cnt*6+3]=bxx[oi*4+3];
        o[cnt*6+4]=sc[oi];      o[cnt*6+5]=1.f;
        ++cnt;
      }
    }
    for (; cnt < DETS; ++cnt){
      o[cnt*6+0]=0.f; o[cnt*6+1]=0.f; o[cnt*6+2]=0.f;
      o[cnt*6+3]=0.f; o[cnt*6+4]=0.f; o[cnt*6+5]=0.f;
    }
  }
}

// ================================================================ launch
extern "C" void kernel_launch(void* const* d_in, const int* in_sizes, int n_in,
                              void* d_out, int out_size, void* d_ws, size_t ws_size,
                              hipStream_t stream){
  (void)in_sizes; (void)n_in; (void)out_size; (void)ws_size;
  const float* feat0      = (const float*)d_in[0];
  const float* feat1      = (const float*)d_in[1];
  const float* feat2      = (const float*)d_in[2];
  const float* rpn_conv_w = (const float*)d_in[3];
  const float* rpn_conv_b = (const float*)d_in[4];
  const float* rpn_cls_w  = (const float*)d_in[5];
  const float* rpn_cls_b  = (const float*)d_in[6];
  const float* rpn_box_w  = (const float*)d_in[7];
  const float* rpn_box_b  = (const float*)d_in[8];
  const float* fc1_w      = (const float*)d_in[9];
  const float* fc1_b      = (const float*)d_in[10];
  const float* fc2_w      = (const float*)d_in[11];
  const float* fc2_b      = (const float*)d_in[12];
  const float* cls_w      = (const float*)d_in[13];
  const float* cls_b      = (const float*)d_in[14];
  const float* box_w      = (const float*)d_in[15];
  const float* box_b      = (const float*)d_in[16];
  float* out = (float*)d_out;

  char* ws = (char*)d_ws;
  size_t off = 0;
  auto alloc = [&](size_t bytes) -> void* {
    void* p = ws + off;
    off = (off + bytes + 255) & ~(size_t)255;
    return p;
  };
  _Float16* W16conv = (_Float16*)alloc((size_t)256*2304*2);      // [cout][2304] f16
  _Float16* Bcol    = (_Float16*)alloc((size_t)2*10016*2304*2);  // im2col, reused per level
  _Float16* headW   = (_Float16*)alloc((size_t)HEADN*256*2);
  float*    headC   = (float*)   alloc((size_t)MPAD32*HEADN*4);
  _Float16* t16     = (_Float16*)alloc((size_t)2*NPOS*256*2);
  float*    objs    = (float*)   alloc((size_t)2*NANCH*4);
  float*    dels    = (float*)   alloc((size_t)2*NANCH*4*4);
  float*    scratch = (float*)   alloc((size_t)2*NANCH*4);
  float*    cscore  = (float*)   alloc((size_t)2*CAND*4);
  float*    cbox    = (float*)   alloc((size_t)2*CAND*4*4);
  float*    props   = (float*)   alloc((size_t)2*POST*4*4);
  _Float16* x16     = (_Float16*)alloc((size_t)1024*FCK*2);
  _Float16* w16fc1t = (_Float16*)alloc((size_t)1024*FCK*2);      // [N=1024][K=12544]
  _Float16* w16fc2t = (_Float16*)alloc((size_t)1024*1024*2);
  _Float16* h1      = (_Float16*)alloc((size_t)1024*1024*2);
  _Float16* h2      = (_Float16*)alloc((size_t)1024*1024*2);
  float*    clsf    = (float*)   alloc((size_t)1024*2*4);
  float*    dltf    = (float*)   alloc((size_t)1024*8*4);

  // 1) weight conversions
  k_cvt_f16<<<(256*2304 + 255)/256, 256, 0, stream>>>(rpn_conv_w, W16conv, 256*2304);
  k_build_headw<<<(HEADN*256 + 255)/256, 256, 0, stream>>>(rpn_cls_w, rpn_box_w, headW);
  k_cvt_t<<<((FCK*1024) + 255)/256, 256, 0, stream>>>(fc1_w, w16fc1t, FCK, 1024);
  k_cvt_t<<<((1024*1024) + 255)/256, 256, 0, stream>>>(fc2_w, w16fc2t, 1024, 1024);

  // 2) RPN 3x3 conv per level: im2col -> WMMA NT GEMM (transposed store into t16)
  const int   Ssz[3]    = {100, 50, 25};
  const int   posOff[3] = {0, 10000, 12500};
  const float* feats[3] = {feat0, feat1, feat2};
  for (int lvl = 0; lvl < 3; ++lvl){
    const int S = Ssz[lvl];
    const int HW = S*S;
    const int HWpad = (HW + 31) & ~31;
    const int tot = 2*HWpad*256;
    k_im2col<<<(tot + 255)/256, 256, 0, stream>>>(feats[lvl], Bcol, S, HWpad);
    dim3 grid(HWpad/32, 256/32, 2);   // 32x32 tile per wave
    k_gemm_nt<<<grid, 32, 0, stream>>>(W16conv, Bcol, rpn_conv_b, t16, nullptr,
                                       256, HWpad, 2304, /*mode=*/1, HW, posOff[lvl]);
  }

  // 3) 1x1 RPN heads as WMMA GEMM: t16 [26250x256] @ headW^T [96x256] -> headC f32
  {
    dim3 grid(HEADN/32, MPAD32/32, 1);
    k_gemm_nt<<<grid, 32, 0, stream>>>(t16, headW, nullptr, nullptr, headC,
                                       MROWS, HEADN, 256, /*mode=*/2, 0, 0);
    k_split_heads<<<(MROWS*75 + 255)/256, 256, 0, stream>>>(headC, rpn_cls_b, rpn_box_b,
                                                            objs, dels);
  }

  // 4) per-level top-500 + anchor decode
  k_topk_decode<<<dim3(3, 2), 1024, 0, stream>>>(objs, dels, scratch, cscore, cbox);

  // 5) RPN sort + batched NMS -> 512 proposals per image
  k_rpn_nms<<<2, 1024, 0, stream>>>(cscore, cbox, props);

  // 6) ROI align -> x16
  k_roi<<<50176, 256, 0, stream>>>(feat0, feat1, feat2, props, x16);

  // 7) fc1 + relu, fc2 + relu (WMMA NT GEMMs, 32x32 tiles)
  k_gemm_nt<<<dim3(1024/32, 1024/32, 1), 32, 0, stream>>>(x16, w16fc1t, fc1_b, h1, nullptr,
                                                          1024, 1024, FCK, /*mode=*/0, 0, 0);
  k_gemm_nt<<<dim3(1024/32, 1024/32, 1), 32, 0, stream>>>(h1, w16fc2t, fc2_b, h2, nullptr,
                                                          1024, 1024, 1024, /*mode=*/0, 0, 0);

  // 8) cls/box heads
  k_cls_box<<<4, 256, 0, stream>>>(h2, cls_w, cls_b, box_w, box_b, clsf, dltf);

  // 9) postprocess -> [2,100,6]
  k_post<<<2, POST, 0, stream>>>(clsf, dltf, props, out);
}
